// TopKRouter_53695681135038
// MI455X (gfx1250) — matrix operand order repaired and checked by reading, not compiled, not run on hardware
//
#include <hip/hip_runtime.h>
#include <hip/hip_bf16.h>
#include <math.h>

#define N_TOKENS  16384
#define IN_FEAT   4096
#define N_EXPERTS 64
#define TOP_K     2
#define CAPACITY  640

typedef __attribute__((ext_vector_type(16))) __bf16 v16bf;
typedef __attribute__((ext_vector_type(8)))  float  v8f;
typedef __attribute__((ext_vector_type(4)))  float  v4f;

// ---------------------------------------------------------------------------
// Init: zero final_probs, set final_indices to INT32_MAX, zero expert counters.
// ---------------------------------------------------------------------------
__global__ void moe_init(float* __restrict__ out_p, int* __restrict__ out_i,
                         int* __restrict__ slots) {
  int i = blockIdx.x * blockDim.x + threadIdx.x;
  if (i < N_TOKENS * TOP_K) {
    out_p[i] = 0.0f;
    out_i[i] = 0x7fffffff;  // INT32_MAX = dropped
  }
  if (i < N_EXPERTS) slots[i] = 0;
}

// ---------------------------------------------------------------------------
// Convert W (f32) to split bf16: W = Whi + Wlo  (error ~2^-16 relative)
// ---------------------------------------------------------------------------
__global__ void moe_cvt_w(const float* __restrict__ W, __bf16* __restrict__ Whi,
                          __bf16* __restrict__ Wlo) {
  int i = blockIdx.x * blockDim.x + threadIdx.x;  // exactly 64*4096 threads
  float w  = W[i];
  __bf16 h = (__bf16)w;
  Whi[i] = h;
  Wlo[i] = (__bf16)(w - (float)h);
}

// ---------------------------------------------------------------------------
// Split-bf16 conversion of a 16-element A fragment.
// ---------------------------------------------------------------------------
__device__ __forceinline__ void cvt_split16(const v4f a[4], v16bf& hi, v16bf& lo) {
#pragma unroll
  for (int q = 0; q < 4; ++q) {
#pragma unroll
    for (int i = 0; i < 4; ++i) {
      float  f = a[q][i];
      __bf16 h = (__bf16)f;
      hi[q * 4 + i] = h;
      lo[q * 4 + i] = (__bf16)(f - (float)h);
    }
  }
}

// A-fragment column offsets within a 32-wide K chunk (contiguous order, matches B):
// lane<16 holds K 0-15, lane>=16 holds K 16-31 (half*16 baked in).
#define ACO(i) ((i) * 4)

// ---------------------------------------------------------------------------
// Split-K GEMM: logits_part[split][N,64] = x[:, krange] @ W[:, krange]^T.
// Block = 256 threads = 8 waves; each wave owns 32 tokens x 64 experts
// (2 M-tiles x 4 N-tiles of 16x16 f32 accumulators). K chunk = 32.
// Pipeline order inside the loop: convert current A (consumes prefetch regs),
// issue next chunk's A loads (refill the now-dead regs -> no register
// rotation copies), then the 24-WMMA block under which the loads retire.
// ---------------------------------------------------------------------------
__global__ __launch_bounds__(256)
void moe_gemm_wmma(const float* __restrict__ x, const __bf16* __restrict__ Whi,
                   const __bf16* __restrict__ Wlo, float* __restrict__ logits_part,
                   int kspan) {
  const int lane = threadIdx.x & 31;
  const int wave = threadIdx.x >> 5;
  const int half = lane >> 4;   // 0: lanes 0-15, 1: lanes 16-31
  const int mrow = lane & 15;
  const int tok0 = blockIdx.x * 256 + wave * 32;
  const int k0   = blockIdx.y * kspan;
  const int kend = k0 + kspan;

  const float* xrow0 = x + (size_t)(tok0 + mrow) * IN_FEAT + half * 16;
  const float* xrow1 = xrow0 + (size_t)16 * IN_FEAT;

  v8f acc[2][4];
#pragma unroll
  for (int u = 0; u < 2; ++u)
#pragma unroll
    for (int t = 0; t < 4; ++t) acc[u][t] = (v8f){};

  // Prefetch first chunk (raw f32).
  v4f pa0[4], pa1[4];
#pragma unroll
  for (int i = 0; i < 4; ++i) {
    pa0[i] = *(const v4f*)(xrow0 + k0 + ACO(i));
    pa1[i] = *(const v4f*)(xrow1 + k0 + ACO(i));
  }

  for (int kc = k0; kc < kend; kc += 32) {
    // Convert current chunk first: pa registers become dead here.
    v16bf ahi0, alo0, ahi1, alo1;
    cvt_split16(pa0, ahi0, alo0);
    cvt_split16(pa1, ahi1, alo1);

    // Refill prefetch registers with next chunk (uniform select; last iter
    // harmlessly re-reads the current, cache-hot chunk).
    const int knext = (kc + 32 < kend) ? (kc + 32) : kc;
#pragma unroll
    for (int i = 0; i < 4; ++i) {
      pa0[i] = *(const v4f*)(xrow0 + knext + ACO(i));
      pa1[i] = *(const v4f*)(xrow1 + knext + ACO(i));
    }

    const size_t wbase = (size_t)mrow * IN_FEAT + (size_t)kc + (size_t)half * 16;
#pragma unroll
    for (int t = 0; t < 4; ++t) {
      const size_t wo  = wbase + (size_t)(t * 16) * IN_FEAT;
      v16bf bhi = *(const v16bf*)(Whi + wo);
      v16bf blo = *(const v16bf*)(Wlo + wo);
      // Interleave the two M-tiles: breaks same-accumulator back-to-back chains,
      // and each B fragment feeds 2 WMMAs (halves W-side fetch per FLOP).
      acc[0][t] = __builtin_amdgcn_wmma_f32_16x16x32_bf16(false, ahi0, false, bhi,
                                                          (short)0, acc[0][t], false, false);
      acc[1][t] = __builtin_amdgcn_wmma_f32_16x16x32_bf16(false, ahi1, false, bhi,
                                                          (short)0, acc[1][t], false, false);
      acc[0][t] = __builtin_amdgcn_wmma_f32_16x16x32_bf16(false, ahi0, false, blo,
                                                          (short)0, acc[0][t], false, false);
      acc[1][t] = __builtin_amdgcn_wmma_f32_16x16x32_bf16(false, ahi1, false, blo,
                                                          (short)0, acc[1][t], false, false);
      acc[0][t] = __builtin_amdgcn_wmma_f32_16x16x32_bf16(false, alo0, false, bhi,
                                                          (short)0, acc[0][t], false, false);
      acc[1][t] = __builtin_amdgcn_wmma_f32_16x16x32_bf16(false, alo1, false, bhi,
                                                          (short)0, acc[1][t], false, false);
    }
  }

  // Store partials: element r of accumulator = row (tok0 + u*16 + r + 8*half), col mrow.
  float* part = logits_part + (size_t)blockIdx.y * N_TOKENS * N_EXPERTS;
#pragma unroll
  for (int u = 0; u < 2; ++u) {
#pragma unroll
    for (int r = 0; r < 8; ++r) {
      float* lrow = part + (size_t)(tok0 + u * 16 + r + half * 8) * N_EXPERTS + mrow;
      lrow[0]  = acc[u][0][r];
      lrow[16] = acc[u][1][r];
      lrow[32] = acc[u][2][r];
      lrow[48] = acc[u][3][r];
    }
  }
}

// ---------------------------------------------------------------------------
// Deterministic split-K reduction + top-2 + softmax-of-2 + scatter into
// per-expert compacted lists; counts assignments per expert.
// ---------------------------------------------------------------------------
__global__ __launch_bounds__(256)
void moe_top2(const float* __restrict__ logits_part, int nsplit,
              int* __restrict__ slots, int* __restrict__ list_tk,
              float* __restrict__ list_p) {
  int t = blockIdx.x * blockDim.x + threadIdx.x;
  if (t >= N_TOKENS) return;

  float v1 = -INFINITY, v2 = -INFINITY;
  int   i1 = 0, i2 = 0;
#pragma unroll 4
  for (int j4 = 0; j4 < N_EXPERTS; j4 += 4) {
    v4f v = {};
    for (int s = 0; s < nsplit; ++s)  // fixed ascending order -> bit-deterministic
      v += *(const v4f*)(logits_part + (size_t)s * N_TOKENS * N_EXPERTS +
                         (size_t)t * N_EXPERTS + j4);
#pragma unroll
    for (int q = 0; q < 4; ++q) {
      float vv = v[q];
      int   j  = j4 + q;
      if (vv > v1)      { v2 = v1; i2 = i1; v1 = vv; i1 = j; }
      else if (vv > v2) { v2 = vv; i2 = j; }
    }
  }
  float e2    = expf(v2 - v1);
  float denom = 1.0f + e2;
  float p1    = 1.0f / denom;
  float p2    = e2 / denom;

  int pos1 = atomicAdd(&slots[i1], 1);
  list_tk[(size_t)i1 * N_TOKENS + pos1] = t * 2 + 0;
  list_p [(size_t)i1 * N_TOKENS + pos1] = p1;
  int pos2 = atomicAdd(&slots[i2], 1);
  list_tk[(size_t)i2 * N_TOKENS + pos2] = t * 2 + 1;
  list_p [(size_t)i2 * N_TOKENS + pos2] = p2;
}

// ---------------------------------------------------------------------------
// Capacity: one block per expert. If count <= 640 keep all; otherwise exact
// rank selection (higher prob wins; tie -> lower token id, matching top_k),
// with the comparison stream staged through LDS in 512-entry chunks.
// ---------------------------------------------------------------------------
__global__ __launch_bounds__(256)
void moe_capacity(const int* __restrict__ slots, const int* __restrict__ list_tk,
                  const float* __restrict__ list_p, float* __restrict__ out_p,
                  int* __restrict__ out_i, float* __restrict__ out_c) {
  __shared__ float sp[512];
  __shared__ int   st[512];

  const int e    = blockIdx.x;
  const int c    = slots[e];
  const int base = e * N_TOKENS;
  if (threadIdx.x == 0) out_c[e] = (float)c;

  if (c <= CAPACITY) {
    for (int i = threadIdx.x; i < c; i += blockDim.x) {
      int tk = list_tk[base + i];
      out_p[tk] = list_p[base + i];
      out_i[tk] = e;
    }
  } else {
    for (int i0 = 0; i0 < c; i0 += blockDim.x) {        // uniform trip count
      const int  i   = i0 + (int)threadIdx.x;
      const bool act = (i < c);
      int   tk  = act ? list_tk[base + i] : 0;
      float p   = act ? list_p[base + i] : 0.0f;
      int   tok = tk >> 1;
      int   rank = 0;
      for (int j0 = 0; j0 < c; j0 += 512) {             // uniform trip count
        const int n = min(512, c - j0);
        __syncthreads();
        for (int j = threadIdx.x; j < n; j += blockDim.x) {
          sp[j] = list_p[base + j0 + j];
          st[j] = list_tk[base + j0 + j] >> 1;
        }
        __syncthreads();
        if (act) {
          for (int j = 0; j < n; ++j)
            rank += (int)((sp[j] > p) | ((sp[j] == p) & (st[j] < tok)));
        }
      }
      if (act && rank < CAPACITY) {
        out_p[tk] = p;
        out_i[tk] = e;
      }
    }
  }
}

// ---------------------------------------------------------------------------
extern "C" void kernel_launch(void* const* d_in, const int* in_sizes, int n_in,
                              void* d_out, int out_size, void* d_ws, size_t ws_size,
                              hipStream_t stream) {
  const float* x = (const float*)d_in[0];  // [16384, 4096] f32
  const float* W = (const float*)d_in[1];  // [64, 4096] f32

  // Fixed workspace needs (everything except split-K partials):
  const size_t sz_logits = (size_t)N_TOKENS * N_EXPERTS * 4;      // 4 MB per split
  const size_t sz_whalf  = (size_t)N_EXPERTS * IN_FEAT * 2;       // 512 KB each
  const size_t sz_lists  = (size_t)N_EXPERTS * N_TOKENS * 4;      // 4 MB each
  const size_t sz_fixed  = 2 * sz_whalf + 1024 + 2 * sz_lists;

  // Pick split-K factor for device-wide parallelism, bounded by ws_size.
  int nsplit = 4;
  while (nsplit > 1 && sz_fixed + (size_t)nsplit * sz_logits > ws_size) nsplit >>= 1;
  const int kspan = IN_FEAT / nsplit;

  char* ws = (char*)d_ws;
  size_t off = 0;
  float*  logits_part = (float*)(ws + off); off += (size_t)nsplit * sz_logits;
  __bf16* Whi         = (__bf16*)(ws + off); off += sz_whalf;
  __bf16* Wlo         = (__bf16*)(ws + off); off += sz_whalf;
  int*    slots       = (int*)(ws + off);    off += 1024;
  int*    list_tk     = (int*)(ws + off);    off += sz_lists;
  float*  list_p      = (float*)(ws + off);  off += sz_lists;

  float* out_p = (float*)d_out;                              // [16384, 2] f32
  int*   out_i = (int*)d_out + N_TOKENS * TOP_K;             // [16384, 2] i32 (bit pattern)
  float* out_c = (float*)d_out + 2 * N_TOKENS * TOP_K;       // [64] f32

  moe_init<<<(N_TOKENS * TOP_K + 255) / 256, 256, 0, stream>>>(out_p, out_i, slots);
  moe_cvt_w<<<(N_EXPERTS * IN_FEAT) / 256, 256, 0, stream>>>(W, Whi, Wlo);

  dim3 ggrid(N_TOKENS / 256, nsplit);
  moe_gemm_wmma<<<ggrid, 256, 0, stream>>>(x, Whi, Wlo, logits_part, kspan);

  moe_top2<<<N_TOKENS / 256, 256, 0, stream>>>(logits_part, nsplit, slots, list_tk, list_p);
  moe_capacity<<<N_EXPERTS, 256, 0, stream>>>(slots, list_tk, list_p, out_p, out_i, out_c);
}
